// CustomTransformer_50062138802561
// MI455X (gfx1250) — matrix operand order repaired and checked by reading, not compiled, not run on hardware
//
#include <hip/hip_runtime.h>
#include <hip/hip_bf16.h>

typedef __attribute__((ext_vector_type(16))) _Float16 v16h;
typedef __attribute__((ext_vector_type(8)))  float    v8f;

#define LN_EPS 1e-5f

// ---------------------------------------------------------------------------
// B-tile loader: 32x16 f16 fragment, K-range [k0, k0+32)
//   BT=1 -> B stored [N,K] (weight layout), BT=0 -> B stored [K,N]
//   B order used here (same as A): lane n = lane&15, element e -> K = k0 + (lane>>4)*16 + e
// ---------------------------------------------------------------------------
template <int BT>
__device__ __forceinline__ v16h load_b_tile(const float* __restrict__ Bb,
                                            int ldb, int bcol, int k0, int h)
{
    v16h b;
    if (BT) {
        const float* bp = Bb + (long long)bcol * ldb + k0 + h * 16;
#pragma unroll
        for (int e = 0; e < 16; ++e) b[e] = (_Float16)bp[e];
    } else {
        const float* bp = Bb + (long long)(k0 + h * 16) * ldb + bcol;
#pragma unroll
        for (int e = 0; e < 16; ++e) b[e] = (_Float16)bp[(long long)e * ldb];
    }
    return b;
}

// ---------------------------------------------------------------------------
// Generic batched GEMM:  C = act( scale * (A @ op(B)) + bias )
//   One wave computes a 16x64 strip of C (4 accumulators) so the A fragment
//   is reused 4x; K advances by 64 per iteration -> 8 WMMAs per loop body.
//   BT compiled in so the inner loop is branch-free (no acc register churn).
//   Requires M%16==0, N%64==0, K%64==0 (true for every call in this model).
// ---------------------------------------------------------------------------
template <int BT>
__global__ __launch_bounds__(256) void wmma_gemm_f16(
    const float* __restrict__ A, int lda, long long sA1, long long sA2,
    const float* __restrict__ B, int ldb, long long sB1, long long sB2,
    float* __restrict__ C, int ldc, long long sC1, long long sC2,
    const float* __restrict__ bias, int relu, float scale,
    int M, int N, int K, int nTiles, int nbi)
{
    const int wave = (int)((blockIdx.x * blockDim.x + threadIdx.x) >> 5);
    const int lane = (int)(threadIdx.x & 31);
    if (wave >= nTiles) return;                 // wave-uniform: EXEC stays all-1s

    const int tilesM   = M >> 4;
    const int tilesN   = N >> 6;                // 64-wide strips
    const int tilesPer = tilesM * tilesN;
    const int batch = wave / tilesPer;
    const int tid2  = wave % tilesPer;
    const int tm = tid2 / tilesN;
    const int tn = tid2 % tilesN;
    const int b1 = batch / nbi;
    const int b2 = batch % nbi;

    const float* Ab = A + (long long)b1 * sA1 + (long long)b2 * sA2;
    const float* Bb = B + (long long)b1 * sB1 + (long long)b2 * sB2;
    float*       Cb = C + (long long)b1 * sC1 + (long long)b2 * sC2;

    const int m = lane & 15;        // row within tile (A) / col within tile (B,C)
    const int h = lane >> 4;        // lane-half selector

    const int arow  = tm * 16 + m;
    const int cbase = tn * 64 + m;  // first of 4 column sub-tiles for this lane

    v8f acc[4] = {v8f{}, v8f{}, v8f{}, v8f{}};

    for (int kk = 0; kk < K; kk += 64) {
        // A fragments for K [kk,kk+32) and [kk+32,kk+64):
        // lanes 0-15 hold K {0..15}, lanes 16-31 hold K {16..31}
        const float* ap = Ab + (long long)arow * lda + kk;
        if (kk + 64 < K) __builtin_prefetch(ap + 64, 0, 3);
        v16h a0, a1;
#pragma unroll
        for (int e = 0; e < 8; ++e) {
            a0[e]     = (_Float16)ap[h * 16 + e];
            a0[8 + e] = (_Float16)ap[h * 16 + 8 + e];
            a1[e]     = (_Float16)ap[32 + h * 16 + e];
            a1[8 + e] = (_Float16)ap[32 + h * 16 + 8 + e];
        }
#pragma unroll
        for (int n = 0; n < 4; ++n) {
            v16h b0 = load_b_tile<BT>(Bb, ldb, cbase + n * 16, kk,      h);
            v16h b1 = load_b_tile<BT>(Bb, ldb, cbase + n * 16, kk + 32, h);
            acc[n] = __builtin_amdgcn_wmma_f32_16x16x32_f16(false, a0, false, b0,
                                                            (short)0, acc[n], false, false);
            acc[n] = __builtin_amdgcn_wmma_f32_16x16x32_f16(false, a1, false, b1,
                                                            (short)0, acc[n], false, false);
        }
    }

    // C/D layout: VGPR j, lane l -> m = (l>>4)*8 + j, n = l&15
#pragma unroll
    for (int n = 0; n < 4; ++n) {
        const int cn = tn * 64 + n * 16 + (lane & 15);
        const float bv = bias ? bias[cn] : 0.0f;
#pragma unroll
        for (int j = 0; j < 8; ++j) {
            const int cm = tm * 16 + h * 8 + j;
            float v = acc[n][j] * scale + bv;
            if (relu) v = fmaxf(v, 0.0f);
            Cb[(long long)cm * ldc + cn] = v;
        }
    }
}

// ---------------------------------------------------------------------------
// LayerNorm over rows of width n:  out = LN(x + res) * g + b   (res optional)
// ---------------------------------------------------------------------------
__global__ __launch_bounds__(256) void ln_kernel(
    const float* __restrict__ x, const float* __restrict__ res,
    const float* __restrict__ g, const float* __restrict__ b,
    float* __restrict__ out, int n)
{
    const int row = blockIdx.x;
    const int tid = threadIdx.x;
    const float* xr = x + (long long)row * n;
    const float* rr = res ? res + (long long)row * n : nullptr;
    __shared__ float red[256];

    float s = 0.f;
    for (int i = tid; i < n; i += 256) s += xr[i] + (rr ? rr[i] : 0.f);
    red[tid] = s; __syncthreads();
    for (int off = 128; off > 0; off >>= 1) {
        if (tid < off) red[tid] += red[tid + off];
        __syncthreads();
    }
    const float mu = red[0] / (float)n;
    __syncthreads();

    float s2 = 0.f;
    for (int i = tid; i < n; i += 256) {
        float v = xr[i] + (rr ? rr[i] : 0.f) - mu;
        s2 += v * v;
    }
    red[tid] = s2; __syncthreads();
    for (int off = 128; off > 0; off >>= 1) {
        if (tid < off) red[tid] += red[tid + off];
        __syncthreads();
    }
    const float rstd = rsqrtf(red[0] / (float)n + LN_EPS);
    __syncthreads();

    for (int i = tid; i < n; i += 256) {
        float v = xr[i] + (rr ? rr[i] : 0.f);
        out[(long long)row * n + i] = (v - mu) * rstd * g[i] + b[i];
    }
}

// ---------------------------------------------------------------------------
// Row softmax (in place), n = 64, one thread per row (rows are L2-resident).
// ---------------------------------------------------------------------------
__global__ __launch_bounds__(256) void softmax_rows(float* __restrict__ s,
                                                    int rows, int n)
{
    int r = blockIdx.x * blockDim.x + threadIdx.x;
    if (r >= rows) return;
    float* p = s + (long long)r * n;
    float mx = -1e30f;
    for (int i = 0; i < n; ++i) mx = fmaxf(mx, p[i]);
    float sum = 0.f;
    for (int i = 0; i < n; ++i) { float e = expf(p[i] - mx); p[i] = e; sum += e; }
    float inv = 1.f / sum;
    for (int i = 0; i < n; ++i) p[i] *= inv;
}

// ---------------------------------------------------------------------------
// Decoder init: tgt = 0; self-attn K/V cache rows = bias (lin(0,w,b)=b).
// grid = B*(T+1) blocks x 128 threads.
// ---------------------------------------------------------------------------
__global__ __launch_bounds__(128) void dec_init(
    float* __restrict__ tgt, float* __restrict__ kc, float* __restrict__ vc,
    const float* __restrict__ sa_qkv_b)
{
    long long idx = (long long)blockIdx.x * 128 + threadIdx.x;
    int i = threadIdx.x;
    tgt[idx] = 0.f;
    kc[idx]  = sa_qkv_b[128 + i];
    vc[idx]  = sa_qkv_b[256 + i];
}

__device__ __forceinline__ void block_ln128(const float* in, float* out,
                                            const float* g, const float* b,
                                            float* red, int tid)
{
    red[tid] = in[tid]; __syncthreads();
    for (int off = 64; off > 0; off >>= 1) {
        if (tid < off) red[tid] += red[tid + off];
        __syncthreads();
    }
    const float mu = red[0] * (1.f / 128.f);
    __syncthreads();
    const float d = in[tid] - mu;
    red[tid] = d * d; __syncthreads();
    for (int off = 64; off > 0; off >>= 1) {
        if (tid < off) red[tid] += red[tid + off];
        __syncthreads();
    }
    const float rstd = rsqrtf(red[0] * (1.f / 128.f) + LN_EPS);
    __syncthreads();
    out[tid] = d * rstd * g[tid] + b[tid];
    __syncthreads();
}

// ---------------------------------------------------------------------------
// One decoder scan step, fused (self-attn + cross-attn + FFN + 3x LN) for
// the single harvested row t. grid = 32 blocks (batch) x 128 threads.
// Writes output row, appends row t+1 to tgt and to the self K/V caches.
// ---------------------------------------------------------------------------
__global__ __launch_bounds__(128) void dec_step(
    int t,
    float* __restrict__ tgt, float* __restrict__ kc, float* __restrict__ vc,
    const float* __restrict__ ck, const float* __restrict__ cv,
    const float* __restrict__ saw, const float* __restrict__ sab,
    const float* __restrict__ sow, const float* __restrict__ sob,
    const float* __restrict__ caw, const float* __restrict__ cab,
    const float* __restrict__ cow, const float* __restrict__ cob,
    const float* __restrict__ f1w, const float* __restrict__ f1b,
    const float* __restrict__ f2w, const float* __restrict__ f2b,
    const float* __restrict__ g1, const float* __restrict__ b1,
    const float* __restrict__ g2, const float* __restrict__ b2,
    const float* __restrict__ g3, const float* __restrict__ b3,
    float* __restrict__ outp)
{
    const int bb  = blockIdx.x;
    const int tid = threadIdx.x;

    __shared__ float x[128], q[128], o[128], x1[128], x2[128];
    __shared__ float sc[8 * 65];
    __shared__ float red[128];
    __shared__ float ff[16];

    x[tid] = tgt[((long long)bb * 65 + t) * 128 + tid];
    __syncthreads();

    // ---- self attention: q projection ----
    {
        const float* w = saw + (long long)tid * 128;   // wq row tid
        float s = sab[tid];
        for (int k2 = 0; k2 < 128; ++k2) s += x[k2] * w[k2];
        q[tid] = s;
    }
    __syncthreads();

    // scores over 65 cached keys, 8 heads of dh=16, scale 1/sqrt(16)
    for (int idx = tid; idx < 8 * 65; idx += 128) {
        int hh = idx / 65, j = idx % 65;
        const float* kr = kc + ((long long)bb * 65 + j) * 128 + hh * 16;
        float s = 0.f;
        for (int d = 0; d < 16; ++d) s += q[hh * 16 + d] * kr[d];
        sc[idx] = s * 0.25f;
    }
    __syncthreads();
    if (tid < 8) {
        float* p = sc + tid * 65;
        float mx = -1e30f;
        for (int j = 0; j < 65; ++j) mx = fmaxf(mx, p[j]);
        float sum = 0.f;
        for (int j = 0; j < 65; ++j) { float e = expf(p[j] - mx); p[j] = e; sum += e; }
        float inv = 1.f / sum;
        for (int j = 0; j < 65; ++j) p[j] *= inv;
    }
    __syncthreads();
    {
        const float* p = sc + (tid >> 4) * 65;
        float s = 0.f;
        for (int j = 0; j < 65; ++j)
            s += p[j] * vc[((long long)bb * 65 + j) * 128 + tid];
        o[tid] = s;
    }
    __syncthreads();
    {   // out proj + residual -> pre-LN (reuse q)
        const float* w = sow + (long long)tid * 128;
        float s = sob[tid];
        for (int d = 0; d < 128; ++d) s += o[d] * w[d];
        q[tid] = x[tid] + s;
    }
    __syncthreads();
    block_ln128(q, x1, g1, b1, red, tid);

    // ---- cross attention over cached mem K/V (64 keys) ----
    {
        const float* w = caw + (long long)tid * 128;   // wq_c row tid
        float s = cab[tid];
        for (int k2 = 0; k2 < 128; ++k2) s += x1[k2] * w[k2];
        q[tid] = s;
    }
    __syncthreads();
    for (int idx = tid; idx < 8 * 64; idx += 128) {
        int hh = idx >> 6, j = idx & 63;
        const float* kr = ck + ((long long)bb * 64 + j) * 128 + hh * 16;
        float s = 0.f;
        for (int d = 0; d < 16; ++d) s += q[hh * 16 + d] * kr[d];
        sc[hh * 65 + j] = s * 0.25f;
    }
    __syncthreads();
    if (tid < 8) {
        float* p = sc + tid * 65;
        float mx = -1e30f;
        for (int j = 0; j < 64; ++j) mx = fmaxf(mx, p[j]);
        float sum = 0.f;
        for (int j = 0; j < 64; ++j) { float e = expf(p[j] - mx); p[j] = e; sum += e; }
        float inv = 1.f / sum;
        for (int j = 0; j < 64; ++j) p[j] *= inv;
    }
    __syncthreads();
    {
        const float* p = sc + (tid >> 4) * 65;
        float s = 0.f;
        for (int j = 0; j < 64; ++j)
            s += p[j] * cv[((long long)bb * 64 + j) * 128 + tid];
        o[tid] = s;
    }
    __syncthreads();
    {
        const float* w = cow + (long long)tid * 128;
        float s = cob[tid];
        for (int d = 0; d < 128; ++d) s += o[d] * w[d];
        q[tid] = x1[tid] + s;
    }
    __syncthreads();
    block_ln128(q, x2, g2, b2, red, tid);

    // ---- FFN 128 -> 16 (relu) -> 128, residual, LN3 ----
    if (tid < 16) {
        const float* w = f1w + (long long)tid * 128;
        float s = f1b[tid];
        for (int k2 = 0; k2 < 128; ++k2) s += x2[k2] * w[k2];
        ff[tid] = fmaxf(s, 0.f);
    }
    __syncthreads();
    {
        const float* w = f2w + (long long)tid * 16;
        float s = f2b[tid];
        for (int j = 0; j < 16; ++j) s += ff[j] * w[j];
        q[tid] = x2[tid] + s;
    }
    __syncthreads();
    block_ln128(q, o, g3, b3, red, tid);   // o = output row

    // emit: d_out[b,t,:] ; tgt[b,t+1,:] ; new self K/V row t+1
    outp[((long long)bb * 64 + t) * 128 + tid] = o[tid];
    tgt[((long long)bb * 65 + t + 1) * 128 + tid] = o[tid];
    {
        const float* wk2 = saw + (long long)(128 + tid) * 128;
        const float* wv2 = saw + (long long)(256 + tid) * 128;
        float sk = sab[128 + tid], sv = sab[256 + tid];
        for (int d = 0; d < 128; ++d) { sk += o[d] * wk2[d]; sv += o[d] * wv2[d]; }
        kc[((long long)bb * 65 + t + 1) * 128 + tid] = sk;
        vc[((long long)bb * 65 + t + 1) * 128 + tid] = sv;
    }
}

// ---------------------------------------------------------------------------
// Host orchestration
// ---------------------------------------------------------------------------
extern "C" void kernel_launch(void* const* d_in, const int* in_sizes, int n_in,
                              void* d_out, int out_size, void* d_ws, size_t ws_size,
                              hipStream_t stream)
{
    const float* src        = (const float*)d_in[0];
    const float* e_qkv_w    = (const float*)d_in[1];
    const float* e_qkv_b    = (const float*)d_in[2];
    const float* e_out_w    = (const float*)d_in[3];
    const float* e_out_b    = (const float*)d_in[4];
    const float* e_ff1_w    = (const float*)d_in[5];
    const float* e_ff1_b    = (const float*)d_in[6];
    const float* e_ff2_w    = (const float*)d_in[7];
    const float* e_ff2_b    = (const float*)d_in[8];
    const float* e_ln1_g    = (const float*)d_in[9];
    const float* e_ln1_b    = (const float*)d_in[10];
    const float* e_ln2_g    = (const float*)d_in[11];
    const float* e_ln2_b    = (const float*)d_in[12];
    const float* fc_w       = (const float*)d_in[13];
    const float* fc_b       = (const float*)d_in[14];
    const float* d_sa_qkv_w = (const float*)d_in[15];
    const float* d_sa_qkv_b = (const float*)d_in[16];
    const float* d_sa_out_w = (const float*)d_in[17];
    const float* d_sa_out_b = (const float*)d_in[18];
    const float* d_ca_qkv_w = (const float*)d_in[19];
    const float* d_ca_qkv_b = (const float*)d_in[20];
    const float* d_ca_out_w = (const float*)d_in[21];
    const float* d_ca_out_b = (const float*)d_in[22];
    const float* d_ff1_w    = (const float*)d_in[23];
    const float* d_ff1_b    = (const float*)d_in[24];
    const float* d_ff2_w    = (const float*)d_in[25];
    const float* d_ff2_b    = (const float*)d_in[26];
    const float* d_ln1_g    = (const float*)d_in[27];
    const float* d_ln1_b    = (const float*)d_in[28];
    const float* d_ln2_g    = (const float*)d_in[29];
    const float* d_ln2_b    = (const float*)d_in[30];
    const float* d_ln3_g    = (const float*)d_in[31];
    const float* d_ln3_b    = (const float*)d_in[32];
    float* outp = (float*)d_out;

    const int Bm = 32, T = 64, DIN = 512, DOUT = 128, H = 8;
    const int R = Bm * T;                  // 2048 rows

    // workspace carve-up (floats)
    float* ws = (float*)d_ws;
    size_t off = 0;
    auto alloc = [&](size_t n) { float* p = ws + off; off += n; return p; };
    float* qkvb = alloc((size_t)R * 3 * DIN);      // [2048,1536]
    float* sbuf = alloc((size_t)Bm * H * T * T);   // [256,64,64]
    float* obuf = alloc((size_t)R * DIN);          // attn concat
    float* pbuf = alloc((size_t)R * DIN);          // proj / ff2 out
    float* hbuf = alloc((size_t)R * 64);           // ff1 out
    float* xA   = alloc((size_t)R * DIN);
    float* xB   = alloc((size_t)R * DIN);
    float* memf = alloc((size_t)R * DOUT);
    float* ckb  = alloc((size_t)R * DOUT);
    float* cvb  = alloc((size_t)R * DOUT);
    float* tgt  = alloc((size_t)Bm * (T + 1) * DOUT);
    float* kcb  = alloc((size_t)Bm * (T + 1) * DOUT);
    float* vcb  = alloc((size_t)Bm * (T + 1) * DOUT);

    auto gemm = [&](const float* A, int lda, long long sA1, long long sA2,
                    const float* Bp, int ldb, long long sB1, long long sB2, int bt,
                    float* C, int ldc, long long sC1, long long sC2,
                    const float* bias, int relu, float scale,
                    int M, int N, int K, int nbo, int nbi) {
        int nTiles = (M / 16) * (N / 64) * nbo * nbi;   // 16x64 strips per wave
        int blocks = (nTiles + 7) / 8;    // 8 waves (wave32) per 256-thread block
        if (bt)
            wmma_gemm_f16<1><<<dim3(blocks), dim3(256), 0, stream>>>(
                A, lda, sA1, sA2, Bp, ldb, sB1, sB2,
                C, ldc, sC1, sC2, bias, relu, scale, M, N, K, nTiles, nbi);
        else
            wmma_gemm_f16<0><<<dim3(blocks), dim3(256), 0, stream>>>(
                A, lda, sA1, sA2, Bp, ldb, sB1, sB2,
                C, ldc, sC1, sC2, bias, relu, scale, M, N, K, nTiles, nbi);
    };

    // ---------------- encoder ----------------
    const float* cur = src;
    for (int i = 0; i < 4; ++i) {
        const float* qw = e_qkv_w + (size_t)i * 3 * DIN * DIN;
        const float* qb = e_qkv_b + (size_t)i * 3 * DIN;
        // qkv = x @ qkv_w.T + b   [2048,1536]
        gemm(cur, DIN, 0, 0, qw, DIN, 0, 0, 1,
             qkvb, 3 * DIN, 0, 0, qb, 0, 1.0f, R, 3 * DIN, DIN, 1, 1);
        // scores = Q K^T / 8 per (b,h)   [256][64,64]
        gemm(qkvb, 3 * DIN, (long long)T * 3 * DIN, 64,
             qkvb + DIN, 3 * DIN, (long long)T * 3 * DIN, 64, 1,
             sbuf, T, (long long)H * T * T, (long long)T * T,
             nullptr, 0, 0.125f, T, T, 64, Bm, H);
        softmax_rows<<<(Bm * H * T + 255) / 256, 256, 0, stream>>>(sbuf, Bm * H * T, T);
        // O = P @ V per (b,h) -> concat heads into [2048,512]
        gemm(sbuf, T, (long long)H * T * T, (long long)T * T,
             qkvb + 2 * DIN, 3 * DIN, (long long)T * 3 * DIN, 64, 0,
             obuf, DIN, (long long)T * DIN, 64,
             nullptr, 0, 1.0f, T, T, T, Bm, H);
        // out proj
        gemm(obuf, DIN, 0, 0, e_out_w + (size_t)i * DIN * DIN, DIN, 0, 0, 1,
             pbuf, DIN, 0, 0, e_out_b + (size_t)i * DIN, 0, 1.0f, R, DIN, DIN, 1, 1);
        ln_kernel<<<R, 256, 0, stream>>>(pbuf, cur, e_ln1_g + (size_t)i * DIN,
                                         e_ln1_b + (size_t)i * DIN, xA, DIN);
        // FFN 512 -> 64 (relu) -> 512
        gemm(xA, DIN, 0, 0, e_ff1_w + (size_t)i * 64 * DIN, DIN, 0, 0, 1,
             hbuf, 64, 0, 0, e_ff1_b + (size_t)i * 64, 1, 1.0f, R, 64, DIN, 1, 1);
        gemm(hbuf, 64, 0, 0, e_ff2_w + (size_t)i * DIN * 64, 64, 0, 0, 1,
             pbuf, DIN, 0, 0, e_ff2_b + (size_t)i * DIN, 0, 1.0f, R, DIN, 64, 1, 1);
        ln_kernel<<<R, 256, 0, stream>>>(pbuf, xA, e_ln2_g + (size_t)i * DIN,
                                         e_ln2_b + (size_t)i * DIN, xB, DIN);
        cur = xB;
    }

    // fc: [2048,512] -> [2048,128]
    gemm(cur, DIN, 0, 0, fc_w, DIN, 0, 0, 1,
         memf, DOUT, 0, 0, fc_b, 0, 1.0f, R, DOUT, DIN, 1, 1);

    // cross-attention K/V caches from mem (static across all 64 steps)
    gemm(memf, DOUT, 0, 0, d_ca_qkv_w + (size_t)DOUT * DOUT, DOUT, 0, 0, 1,
         ckb, DOUT, 0, 0, d_ca_qkv_b + DOUT, 0, 1.0f, R, DOUT, DOUT, 1, 1);
    gemm(memf, DOUT, 0, 0, d_ca_qkv_w + (size_t)2 * DOUT * DOUT, DOUT, 0, 0, 1,
         cvb, DOUT, 0, 0, d_ca_qkv_b + 2 * DOUT, 0, 1.0f, R, DOUT, DOUT, 1, 1);

    // ---------------- decoder scan ----------------
    dec_init<<<Bm * (T + 1), 128, 0, stream>>>(tgt, kcb, vcb, d_sa_qkv_b);
    for (int t = 0; t < T; ++t) {
        dec_step<<<Bm, 128, 0, stream>>>(t, tgt, kcb, vcb, ckb, cvb,
                                         d_sa_qkv_w, d_sa_qkv_b, d_sa_out_w, d_sa_out_b,
                                         d_ca_qkv_w, d_ca_qkv_b, d_ca_out_w, d_ca_out_b,
                                         d_ff1_w, d_ff1_b, d_ff2_w, d_ff2_b,
                                         d_ln1_g, d_ln1_b, d_ln2_g, d_ln2_b,
                                         d_ln3_g, d_ln3_b, outp);
    }
    (void)in_sizes; (void)n_in; (void)out_size; (void)ws_size;
}
